// CannyFilter_33895881900742
// MI455X (gfx1250) — matrix operand order repaired and checked by reading, not compiled, not run on hardware
//
#include <hip/hip_runtime.h>
#include <hip/hip_bf16.h>
#include <math.h>

// Canny on (8,3,1024,1024) fp32.
// Outputs (concat): grad_x, grad_y, grad_magnitude, grad_orientation, thin_edges
// each (8,1,1024,1024).
//
// Stage 1 (fused, one global read of img): channel-mean -> separable 3x3 sobel
//   -> magnitude (computed on 18x18 halo tile in LDS) -> orientation -> NMS
//   -> double threshold. Writes 4 outputs (NT stores) + thin01 map to ws.
// Stage 2 (hysteresis): 3x3 ones*1.25 conv done as a 16x16x32 f16 WMMA
//   (exact for values in {0,0.5,1}) + weak/strong promotion.

typedef __attribute__((ext_vector_type(16))) _Float16 v16h;
typedef __attribute__((ext_vector_type(8)))  float    v8f;

constexpr int IMG_H = 1024;
constexpr int IMG_W = 1024;
constexpr int HW    = IMG_H * IMG_W;         // 1,048,576
constexpr int NB    = 8;
constexpr int PLANE = NB * HW;               // elements per output tensor

// ---------------------------------------------------------------------------
// Stage 1: sobel + magnitude + orientation + NMS + double threshold (fused)
// Block: 256 threads -> one 16x16 output tile. LDS holds 20x20 mean tile and
// 18x18 magnitude tile (1-pixel NMS halo + 1-pixel sobel halo).
// ---------------------------------------------------------------------------
__global__ __launch_bounds__(256)
void canny_stage1(const float* __restrict__ img,
                  float* __restrict__ out,
                  float* __restrict__ thin01)
{
    __shared__ float gs[20][21];   // channel-mean, rows y0-2..y0+17
    __shared__ float mg[18][19];   // magnitude,    rows y0-1..y0+16

    const int tid = threadIdx.x;
    const int tx  = tid & 15;
    const int ty  = tid >> 4;
    const int x0  = blockIdx.x * 16;
    const int y0  = blockIdx.y * 16;
    const int b   = blockIdx.z;

    const float* im0 = img + (size_t)b * 3 * HW;

    // load 20x20 channel-mean tile (zero padded outside image)
    for (int idx = tid; idx < 400; idx += 256) {
        int r  = idx / 20, c = idx % 20;
        int gy = y0 - 2 + r, gx = x0 - 2 + c;
        float v = 0.0f;
        if ((unsigned)gy < (unsigned)IMG_H && (unsigned)gx < (unsigned)IMG_W) {
            int p = gy * IMG_W + gx;
            v = (im0[p] + im0[p + HW] + im0[p + 2 * HW]) * (1.0f / 3.0f);
        }
        gs[r][c] = v;
    }
    __syncthreads();

    // magnitude on 18x18 region (zero outside image, matching SAME padding of
    // the magnitude tensor used by the directional convs)
    for (int idx = tid; idx < 324; idx += 256) {
        int r  = idx / 18, c = idx % 18;
        int gy = y0 - 1 + r, gx = x0 - 1 + c;
        float m = 0.0f;
        if ((unsigned)gy < (unsigned)IMG_H && (unsigned)gx < (unsigned)IMG_W) {
            int rr = r + 1, cc = c + 1;
            float sx = 0.5f * (gs[rr-1][cc+1] - gs[rr-1][cc-1])
                     +        (gs[rr  ][cc+1] - gs[rr  ][cc-1])
                     + 0.5f * (gs[rr+1][cc+1] - gs[rr+1][cc-1]);
            float sy = 0.5f * (gs[rr+1][cc-1] - gs[rr-1][cc-1])
                     +        (gs[rr+1][cc  ] - gs[rr-1][cc  ])
                     + 0.5f * (gs[rr+1][cc+1] - gs[rr-1][cc+1]);
            m = sqrtf(sx * sx + sy * sy);
        }
        mg[r][c] = m;
    }
    __syncthreads();

    // per-pixel finalization for the 16x16 interior
    {
        const int rr = ty + 2, cc = tx + 2;   // gs coords of this pixel
        float sx = 0.5f * (gs[rr-1][cc+1] - gs[rr-1][cc-1])
                 +        (gs[rr  ][cc+1] - gs[rr  ][cc-1])
                 + 0.5f * (gs[rr+1][cc+1] - gs[rr+1][cc-1]);
        float sy = 0.5f * (gs[rr+1][cc-1] - gs[rr-1][cc-1])
                 +        (gs[rr+1][cc  ] - gs[rr-1][cc  ])
                 + 0.5f * (gs[rr+1][cc+1] - gs[rr-1][cc+1]);
        float mag = mg[ty + 1][tx + 1];

        float ang  = atanf(sy / sx) * 114.59155902616465f + 180.0f; // 360/pi
        float ango = roundf(ang * (1.0f / 45.0f)) * 45.0f;
        float pidx = fmodf(ango * (1.0f / 45.0f), 8.0f);            // 0..7

        // directional[i] = mag(center) - mag(neighbor_i); (dy,dx) table
        const int offy[8] = { 0, -1, -1, -1,  0,  1, 1, 1 };
        const int offx[8] = { 1,  1,  0, -1, -1, -1, 0, 1 };

        float thin = mag;
        #pragma unroll
        for (int i = 0; i < 4; ++i) {
            float dp = mag - mg[ty + 1 + offy[i]    ][tx + 1 + offx[i]    ];
            float dn = mag - mg[ty + 1 + offy[i + 4]][tx + 1 + offx[i + 4]];
            bool is_max = fminf(dp, dn) > 0.0f;
            bool is_or  = (pidx == (float)i) || (pidx == (float)(i + 4));
            if (is_or && !is_max) thin = 0.0f;
        }

        float lo = (thin > 0.5f) ? 0.5f : 0.0f;
        float hi = (thin > 1.0f) ? 0.5f : 0.0f;

        size_t p = (size_t)b * HW + (size_t)(y0 + ty) * IMG_W + (x0 + tx);
        __builtin_nontemporal_store(sx,   out + p);
        __builtin_nontemporal_store(sy,   out + p + (size_t)PLANE);
        __builtin_nontemporal_store(mag,  out + p + 2 * (size_t)PLANE);
        __builtin_nontemporal_store(ango, out + p + 3 * (size_t)PLANE);
        thin01[p] = lo + hi;   // re-read by stage 2: keep cacheable
    }
}

// ---------------------------------------------------------------------------
// Stage 2: hysteresis. Each wave owns a 16x16 tile; the 3x3 ones*1.25 conv is
// D = Aband(16x32,f16) x Hext(32x16,f16) via V_WMMA_F32_16X16X32_F16, exact
// since thin01 in {0,0.5,1}. Block = 8 waves = 128x16 pixel strip.
// ---------------------------------------------------------------------------
__global__ __launch_bounds__(256)
void canny_hyst_wmma(const float* __restrict__ thin01,
                     float* __restrict__ out_thin)
{
    __shared__ float T[130][18];   // strip rows yb-1..yb+128, cols x0-1..x0+16

    const int tid = threadIdx.x;
    const int x0  = blockIdx.x * 16;
    const int yb  = blockIdx.y * 128;
    const int b   = blockIdx.z;
    const float* tp = thin01 + (size_t)b * HW;

    for (int idx = tid; idx < 130 * 18; idx += 256) {
        int r  = idx / 18, c = idx % 18;
        int gy = yb - 1 + r, gx = x0 - 1 + c;
        T[r][c] = ((unsigned)gy < (unsigned)IMG_H && (unsigned)gx < (unsigned)IMG_W)
                      ? tp[gy * IMG_W + gx] : 0.0f;
    }
    __syncthreads();

    const int lane = tid & 31;
    const int wv   = tid >> 5;       // wave -> 16-row sub-tile
    const int n    = lane & 15;      // column within tile / matrix row for A
    const int h    = lane >> 4;      // lane half

    // B operand: Hext[k][n] = horizontal 3-sum at (row y0-1+k, col x0+n);
    // B k order used here (contiguous, same as A): VGPR v holds K = 16*h + 2*v + {0,1}, lane n.
    v16h bm;
    #pragma unroll
    for (int v = 0; v < 8; ++v) {
        #pragma unroll
        for (int s = 0; s < 2; ++s) {
            int k = 16 * h + 2 * v + s;
            float hv = 0.0f;
            if (k < 18) {
                int r = wv * 16 + k;
                hv = T[r][n] + T[r][n + 1] + T[r][n + 2];
            }
            bm[2 * v + s] = (_Float16)hv;
        }
    }

    // A operand: band matrix A[row][K] = (row <= K <= row+2), row = lane&15.
    // A k order used here (contiguous, same as B): VGPR v holds K = 16*h + 2*v + {0,1}.
    v16h am;
    #pragma unroll
    for (int v = 0; v < 8; ++v) {
        #pragma unroll
        for (int s = 0; s < 2; ++s) {
            int K = 16 * h + 2 * v + s;
            am[2 * v + s] = (_Float16)((K >= n && K <= n + 2) ? 1.0f : 0.0f);
        }
    }

    v8f acc = {};
    acc = __builtin_amdgcn_wmma_f32_16x16x32_f16(
        /*neg_a=*/false, am, /*neg_b=*/false, bm,
        /*c_mod=*/(short)0, acc, /*reuse_a=*/false, /*reuse_b=*/false);

    // D layout: lanes 0-15 -> M = vgpr, lanes 16-31 -> M = vgpr+8; N = lane&15.
    const int y0 = yb + wv * 16;
    #pragma unroll
    for (int r = 0; r < 8; ++r) {
        int   M = r + 8 * h;
        float S = acc[r] * 1.25f;                 // conv(thin01, 1.25*ones3x3)
        float t = T[wv * 16 + 1 + M][n + 1];      // thin01 at center
        float o = (t > 0.75f) ? 1.0f
                : ((t > 0.25f && S > 1.0f) ? 1.0f : 0.0f);
        __builtin_nontemporal_store(
            o, out_thin + (size_t)b * HW + (size_t)(y0 + M) * IMG_W + (x0 + n));
    }
}

// ---------------------------------------------------------------------------
extern "C" void kernel_launch(void* const* d_in, const int* in_sizes, int n_in,
                              void* d_out, int out_size, void* d_ws, size_t ws_size,
                              hipStream_t stream)
{
    const float* img = (const float*)d_in[0];
    float* out    = (float*)d_out;
    float* thin01 = (float*)d_ws;   // PLANE floats (33.5 MB) of scratch

    dim3 g1(IMG_W / 16, IMG_H / 16, NB);   // 64 x 64 x 8
    canny_stage1<<<g1, dim3(256), 0, stream>>>(img, out, thin01);

    dim3 g2(IMG_W / 16, IMG_H / 128, NB);  // 64 x 8 x 8
    canny_hyst_wmma<<<g2, dim3(256), 0, stream>>>(thin01,
                                                  out + 4 * (size_t)PLANE);
}
